// TFCvtSelfAttention_11355893530978
// MI455X (gfx1250) — compile-verified, matrix operand order checked by reading, not run on hardware
//
#include <hip/hip_runtime.h>

typedef __attribute__((ext_vector_type(16))) _Float16 v16h;
typedef __attribute__((ext_vector_type(8)))  float    v8f;

#define NB    64
#define CDIM  384
#define NHEAD 6
#define HD    64
#define LQ    785
#define LKV   197
#define HWD   28
#define EPS_F 1e-5f
#define QSPLIT 2

__device__ __forceinline__ v8f wmma16(v16h a, v16h b, v8f c) {
  // D = A(16x32 f16) * B(32x16 f16) + C(16x16 f32)
  return __builtin_amdgcn_wmma_f32_16x16x32_f16(false, a, false, b, (short)0, c,
                                                false, false);
}

// ---------------- stage 1: depthwise conv 3x3 + BN -> f16 ----------------
__global__ void conv_bn_q_kernel(const float* __restrict__ hs,
                                 const float* __restrict__ w,
                                 const float* __restrict__ g,
                                 const float* __restrict__ be,
                                 const float* __restrict__ mu,
                                 const float* __restrict__ va,
                                 _Float16* __restrict__ o) {
  const int b = blockIdx.x / LQ, p = blockIdx.x % LQ, c = threadIdx.x;
  const float* base = hs + (size_t)b * LQ * CDIM;
  if (p == 0) { o[(size_t)b * LQ * CDIM + c] = (_Float16)base[c]; return; }
  const int idx = p - 1, y = idx / HWD, x = idx % HWD;
  float acc = 0.f;
#pragma unroll
  for (int dy = 0; dy < 3; ++dy) {
#pragma unroll
    for (int dx = 0; dx < 3; ++dx) {
      const int iy = y + dy - 1, ix = x + dx - 1;
      if (iy >= 0 && iy < HWD && ix >= 0 && ix < HWD)
        acc += base[(size_t)(1 + iy * HWD + ix) * CDIM + c] *
               w[(dy * 3 + dx) * CDIM + c];
    }
  }
  const float val = (acc - mu[c]) * rsqrtf(va[c] + EPS_F) * g[c] + be[c];
  o[((size_t)b * LQ + p) * CDIM + c] = (_Float16)val;
}

__global__ void conv_bn_kv_kernel(const float* __restrict__ hs,
    const float* __restrict__ wk, const float* __restrict__ gk,
    const float* __restrict__ bek, const float* __restrict__ muk,
    const float* __restrict__ vak,
    const float* __restrict__ wv, const float* __restrict__ gv,
    const float* __restrict__ bev, const float* __restrict__ muv,
    const float* __restrict__ vav,
    _Float16* __restrict__ ok, _Float16* __restrict__ ov) {
  const int b = blockIdx.x / LKV, p = blockIdx.x % LKV, c = threadIdx.x;
  const float* base = hs + (size_t)b * LQ * CDIM;
  if (p == 0) {
    const _Float16 cls = (_Float16)base[c];
    ok[(size_t)b * LKV * CDIM + c] = cls;
    ov[(size_t)b * LKV * CDIM + c] = cls;
    return;
  }
  const int idx = p - 1, y = (idx / 14) * 2, x = (idx % 14) * 2;
  float ak = 0.f, av = 0.f;
#pragma unroll
  for (int dy = 0; dy < 3; ++dy) {
#pragma unroll
    for (int dx = 0; dx < 3; ++dx) {
      const int iy = y + dy - 1, ix = x + dx - 1;
      if (iy >= 0 && iy < HWD && ix >= 0 && ix < HWD) {
        const float xin = base[(size_t)(1 + iy * HWD + ix) * CDIM + c];
        ak += xin * wk[(dy * 3 + dx) * CDIM + c];
        av += xin * wv[(dy * 3 + dx) * CDIM + c];
      }
    }
  }
  const float rk = (ak - muk[c]) * rsqrtf(vak[c] + EPS_F) * gk[c] + bek[c];
  const float rv = (av - muv[c]) * rsqrtf(vav[c] + EPS_F) * gv[c] + bev[c];
  ok[((size_t)b * LKV + p) * CDIM + c] = (_Float16)rk;
  ov[((size_t)b * LKV + p) * CDIM + c] = (_Float16)rv;
}

// ---------------- stage 2: weight downconvert ----------------
__global__ void cvt_w_kernel(const float* __restrict__ a,
                             const float* __restrict__ b,
                             const float* __restrict__ c,
                             _Float16* __restrict__ oa,
                             _Float16* __restrict__ ob,
                             _Float16* __restrict__ oc) {
  const int i = blockIdx.x * blockDim.x + threadIdx.x;
  if (i < CDIM * CDIM) {
    oa[i] = (_Float16)a[i];
    ob[i] = (_Float16)b[i];
    oc[i] = (_Float16)c[i];
  }
}

// ---------------- stage 3: projection GEMM (M x 384) @ (384 x 384) ----------------
// Block: 256 threads = 8 waves, tile 128(M) x 64(N), K chunks of 32,
// double-buffered LDS (one barrier per chunk, loads overlap WMMAs).
template <int L>
__global__ void proj_gemm_kernel(const _Float16* __restrict__ X,
                                 const _Float16* __restrict__ W,
                                 const float* __restrict__ bias,
                                 _Float16* __restrict__ O) {
  constexpr int M = NB * L;
  __shared__ _Float16 sA[2][128 * 32];   // row-major, ld = 32
  __shared__ _Float16 sBt[2][64 * 32];   // transposed: sBt[n][k], ld = 32
  const int tid = threadIdx.x, wave = tid >> 5, lane = tid & 31;
  const int n16 = lane & 15, hi8 = (lane >> 4) << 3;
  const int m0 = blockIdx.x * 128, n0 = blockIdx.y * 64;

  // A-loader: each thread owns one (row, 8-dword half) — clamp hoisted.
  const int arow = tid >> 1, acol8 = (tid & 1) * 8;  // dword units
  int gr = m0 + arow; if (gr >= M) gr = M - 1;
  const uint32_t* Arow32 = (const uint32_t*)X + (size_t)gr * (CDIM / 2);
  // B-loader: each thread owns 8 contiguous halves of one k-row.
  const int brow = tid >> 3, bcol8 = (tid & 7) * 8;  // half units

  auto load_tile = [&](int kk, int buf) {
    const uint4 a0 = *(const uint4*)(Arow32 + kk / 2 + acol8);
    const uint4 a1 = *(const uint4*)(Arow32 + kk / 2 + acol8 + 4);
    uint32_t* dst = (uint32_t*)sA[buf];
    *(uint4*)(dst + arow * 16 + acol8)     = a0;
    *(uint4*)(dst + arow * 16 + acol8 + 4) = a1;
    union { uint4 u; _Float16 h[8]; } wv;
    wv.u = *(const uint4*)(W + (size_t)(kk + brow) * CDIM + n0 + bcol8);
#pragma unroll
    for (int e = 0; e < 8; ++e) sBt[buf][(bcol8 + e) * 32 + brow] = wv.h[e];
  };

  v8f acc[4] = {};
  load_tile(0, 0);
  __syncthreads();
  int pb = 0;
  for (int kk = 0; kk < CDIM; kk += 32) {
    if (kk + 32 < CDIM) load_tile(kk + 32, pb ^ 1);

    v16h a;
#pragma unroll
    for (int j = 0; j < 8; ++j) {
      const int k0 = ((j >> 2) << 4) + hi8 + ((j & 3) << 1);
      a[2 * j]     = sA[pb][(wave * 16 + n16) * 32 + k0];
      a[2 * j + 1] = sA[pb][(wave * 16 + n16) * 32 + k0 + 1];
    }
#pragma unroll
    for (int t = 0; t < 4; ++t) {
      v16h bb;
#pragma unroll
      for (int j = 0; j < 8; ++j) {
        const int k0 = ((j >> 2) << 4) + hi8 + ((j & 3) << 1);
        bb[2 * j]     = sBt[pb][(t * 16 + n16) * 32 + k0];
        bb[2 * j + 1] = sBt[pb][(t * 16 + n16) * 32 + k0 + 1];
      }
      acc[t] = wmma16(a, bb, acc[t]);
    }
    __syncthreads();
    pb ^= 1;
  }

#pragma unroll
  for (int t = 0; t < 4; ++t) {
    const int col = n0 + t * 16 + n16;
    const float bv = bias[col];
    const int h = col >> 6, d = col & 63;
#pragma unroll
    for (int r = 0; r < 8; ++r) {
      const int rm = m0 + wave * 16 + r + hi8;
      if (rm < M) {
        const int bi = rm / L, l = rm % L;   // constant L -> mul/shift
        O[(((size_t)bi * NHEAD + h) * L + l) * HD + d] = (_Float16)(acc[t][r] + bv);
      }
    }
  }
}

// ---------------- stage 4: flash attention ----------------
// grid = (NB*NHEAD, QSPLIT); each block owns every QSPLIT-th group of 8 q-tiles.
// Occupancy is LDS-bound (64KB/block), so relax the register budget to kill
// the scratch spills of the Q fragments across the chunk loop.
__global__ void __launch_bounds__(256, 1)
flash_attn_kernel(const _Float16* __restrict__ Q,
                  const _Float16* __restrict__ K,
                  const _Float16* __restrict__ V,
                  float* __restrict__ out) {
  __shared__ _Float16 sK[224 * 64];   // [key][d]; keys 197..223 masked in S
  __shared__ _Float16 sVt[64 * 224];  // [d][key]; padded keys zeroed
  __shared__ _Float16 sP[8 * 512];    // per-wave 16x32 P staging
  const float scale = 0.05103103630798287f;  // 384^-0.5

  const int tid = threadIdx.x;
  const int b = blockIdx.x / NHEAD, h = blockIdx.x % NHEAD;
  const _Float16* Kb = K + ((size_t)b * NHEAD + h) * LKV * HD;
  const _Float16* Vb = V + ((size_t)b * NHEAD + h) * LKV * HD;
  const _Float16* Qb = Q + ((size_t)b * NHEAD + h) * LQ * HD;

  // K tile: async DMA global -> LDS, 7 sweeps x 256 lanes x 16B = 28,672B.
  // Tail rows read past this slice (still inside workspace); those key
  // columns are overwritten with -1e30 before softmax, so contents are moot.
  {
    const uint32_t lds0 = (uint32_t)(uintptr_t)(&sK[0]) + (uint32_t)tid * 16u;
    const uint64_t g0   = (uint64_t)(uintptr_t)Kb + (uint64_t)tid * 16u;
#pragma unroll
    for (int s = 0; s < 7; ++s) {
      const uint32_t loff = lds0 + (uint32_t)s * 4096u;
      const uint64_t ga   = g0 + (uint64_t)s * 4096u;
      asm volatile("global_load_async_to_lds_b128 %0, %1, off"
                   :: "v"(loff), "v"(ga) : "memory");
    }
  }
  // V tile: vectorized load + transpose into sVt; padded keys must be zero.
  for (int i = tid; i < 224 * 8; i += 256) {
    const int key = i >> 3, db = (i & 7) * 8;
    union { uint4 u; _Float16 h[8]; } vv;
    if (key < LKV) vv.u = *(const uint4*)(Vb + (size_t)key * HD + db);
    else           vv.u = uint4{0u, 0u, 0u, 0u};
#pragma unroll
    for (int e = 0; e < 8; ++e) sVt[(db + e) * 224 + key] = vv.h[e];
  }
  asm volatile("s_wait_asynccnt 0" ::: "memory");
  __syncthreads();

  const int wave = tid >> 5, lane = tid & 31;
  const int n16 = lane & 15, hi8 = (lane >> 4) << 3;

  // Constant "ones-column" B fragment: column 0 (owned by lanes with n16==0
  // in both k-halves) is 1.0; P x bones accumulates the softmax row-sums on
  // the matrix pipe instead of 32 ds_bpermutes per chunk.
  v16h bones;
#pragma unroll
  for (int e = 0; e < 16; ++e) bones[e] = (n16 == 0) ? (_Float16)1.f : (_Float16)0.f;

  for (int tile = blockIdx.y * 8 + wave; tile < 50; tile += 8 * QSPLIT) {
    const int q0 = tile * 16;
    // Q A-fragments: two contiguous 8-half runs per 32-dim block -> b128 loads
    int qrow = q0 + n16; if (qrow > LQ - 1) qrow = LQ - 1;
    const _Float16* qp = Qb + (size_t)qrow * HD;
    union { uint4 u; _Float16 h[8]; } q00, q01, q10, q11;
    q00.u = *(const uint4*)(qp + hi8);        // halves hi8 .. hi8+7
    q01.u = *(const uint4*)(qp + 16 + hi8);   // halves 16+hi8 ..
    q10.u = *(const uint4*)(qp + 32 + hi8);
    q11.u = *(const uint4*)(qp + 48 + hi8);
    v16h aq0, aq1;
#pragma unroll
    for (int e = 0; e < 8; ++e) {
      aq0[e]     = q00.h[e];
      aq0[e + 8] = q01.h[e];
      aq1[e]     = q10.h[e];
      aq1[e + 8] = q11.h[e];
    }

    float mrow[8];
    v8f oacc[4] = {};
    v8f lacc = {};
#pragma unroll
    for (int r = 0; r < 8; ++r) mrow[r] = -1e30f;

    // One key-chunk of 32: 4 S-WMMAs, online softmax, 5 PV-WMMAs.
    // `mask` is a literal at each call site, so chunks 0..5 skip the compares.
    auto chunk_body = [&](int ch, bool mask) {
      v8f s0 = {}, s1 = {};
      {
        const int key = ch * 32 + n16;
        v16h bk;
#pragma unroll
        for (int j = 0; j < 8; ++j) {
          const int k0 = ((j >> 2) << 4) + hi8 + ((j & 3) << 1);
          bk[2 * j] = sK[key * 64 + k0]; bk[2 * j + 1] = sK[key * 64 + k0 + 1];
        }
        s0 = wmma16(aq0, bk, s0);
#pragma unroll
        for (int j = 0; j < 8; ++j) {
          const int k0 = ((j >> 2) << 4) + hi8 + ((j & 3) << 1) + 32;
          bk[2 * j] = sK[key * 64 + k0]; bk[2 * j + 1] = sK[key * 64 + k0 + 1];
        }
        s0 = wmma16(aq1, bk, s0);
      }
      {
        const int key = ch * 32 + 16 + n16;
        v16h bk;
#pragma unroll
        for (int j = 0; j < 8; ++j) {
          const int k0 = ((j >> 2) << 4) + hi8 + ((j & 3) << 1);
          bk[2 * j] = sK[key * 64 + k0]; bk[2 * j + 1] = sK[key * 64 + k0 + 1];
        }
        s1 = wmma16(aq0, bk, s1);
#pragma unroll
        for (int j = 0; j < 8; ++j) {
          const int k0 = ((j >> 2) << 4) + hi8 + ((j & 3) << 1) + 32;
          bk[2 * j] = sK[key * 64 + k0]; bk[2 * j + 1] = sK[key * 64 + k0 + 1];
        }
        s1 = wmma16(aq1, bk, s1);
      }

      // scale (+ padding mask only on the final chunk)
#pragma unroll
      for (int r = 0; r < 8; ++r) { s0[r] *= scale; s1[r] *= scale; }
      if (mask) {
        const int key0 = ch * 32 + n16, key1 = key0 + 16;
#pragma unroll
        for (int r = 0; r < 8; ++r) {
          s0[r] = (key0 < LKV) ? s0[r] : -1e30f;
          s1[r] = (key1 < LKV) ? s1[r] : -1e30f;
        }
      }

      // row max: stage-parallel butterfly (8 independent permutes per stage)
      float mx[8];
#pragma unroll
      for (int r = 0; r < 8; ++r) mx[r] = fmaxf(s0[r], s1[r]);
#pragma unroll
      for (int off = 8; off >= 1; off >>= 1) {
#pragma unroll
        for (int r = 0; r < 8; ++r)
          mx[r] = fmaxf(mx[r], __shfl_xor(mx[r], off, 32));
      }

#pragma unroll
      for (int r = 0; r < 8; ++r) {
        const float mnew = fmaxf(mrow[r], mx[r]);
        const float corr = __expf(mrow[r] - mnew);
        lacc[r] *= corr;
        oacc[0][r] *= corr; oacc[1][r] *= corr;
        oacc[2][r] *= corr; oacc[3][r] *= corr;
        mrow[r] = mnew;
      }
#pragma unroll
      for (int r = 0; r < 8; ++r) {
        const float p0 = __expf(s0[r] - mrow[r]);
        const float p1 = __expf(s1[r] - mrow[r]);
        sP[wave * 512 + (r + hi8) * 32 + n16]      = (_Float16)p0;
        sP[wave * 512 + (r + hi8) * 32 + 16 + n16] = (_Float16)p1;
      }
      asm volatile("s_wait_dscnt 0" ::: "memory");
      __builtin_amdgcn_wave_barrier();

      // P as A-fragment (16x32); O += P*V; l += P*ones (5th WMMA)
      v16h ap;
#pragma unroll
      for (int j = 0; j < 8; ++j) {
        const int k0 = ((j >> 2) << 4) + hi8 + ((j & 3) << 1);
        ap[2 * j]     = sP[wave * 512 + n16 * 32 + k0];
        ap[2 * j + 1] = sP[wave * 512 + n16 * 32 + k0 + 1];
      }
#pragma unroll
      for (int t = 0; t < 4; ++t) {
        v16h bv;
        const int d = t * 16 + n16;
#pragma unroll
        for (int j = 0; j < 8; ++j) {
          const int k0 = ((j >> 2) << 4) + hi8 + ((j & 3) << 1);
          bv[2 * j]     = sVt[d * 224 + ch * 32 + k0];
          bv[2 * j + 1] = sVt[d * 224 + ch * 32 + k0 + 1];
        }
        oacc[t] = wmma16(ap, bv, oacc[t]);
      }
      lacc = wmma16(ap, bones, lacc);
    };

    for (int ch = 0; ch < 6; ++ch) chunk_body(ch, false);  // keys 0..191: no mask
    chunk_body(6, true);                                   // keys 192..223: masked

    // normalize + write fp32 output in (B, Lq, C) layout.
    // Row sums live in lanes with n16==0; broadcast within each 16-lane half.
#pragma unroll
    for (int r = 0; r < 8; ++r) {
      const int row = q0 + r + hi8;
      const float lsum = __shfl(lacc[r], lane & 0x10, 32);
      if (row < LQ) {
        const float inv = 1.f / lsum;
#pragma unroll
        for (int t = 0; t < 4; ++t) {
          const int d = t * 16 + n16;
          out[((size_t)b * LQ + row) * CDIM + h * HD + d] = oacc[t][r] * inv;
        }
      }
    }
  }
}

// ---------------- launch ----------------
extern "C" void kernel_launch(void* const* d_in, const int* in_sizes, int n_in,
                              void* d_out, int out_size, void* d_ws, size_t ws_size,
                              hipStream_t stream) {
  (void)in_sizes; (void)n_in; (void)out_size; (void)ws_size;
  const float* hs  = (const float*)d_in[0];
  const float* wdq = (const float*)d_in[3];
  const float* gq  = (const float*)d_in[4];
  const float* beq = (const float*)d_in[5];
  const float* muq = (const float*)d_in[6];
  const float* vaq = (const float*)d_in[7];
  const float* Wq  = (const float*)d_in[8];
  const float* bq  = (const float*)d_in[9];
  const float* wdk = (const float*)d_in[10];
  const float* gk  = (const float*)d_in[11];
  const float* bek = (const float*)d_in[12];
  const float* muk = (const float*)d_in[13];
  const float* vak = (const float*)d_in[14];
  const float* Wk  = (const float*)d_in[15];
  const float* bk  = (const float*)d_in[16];
  const float* wdv = (const float*)d_in[17];
  const float* gv  = (const float*)d_in[18];
  const float* bev = (const float*)d_in[19];
  const float* muv = (const float*)d_in[20];
  const float* vav = (const float*)d_in[21];
  const float* Wv  = (const float*)d_in[22];
  const float* bv  = (const float*)d_in[23];

  _Float16* ws = (_Float16*)d_ws;
  size_t off = 0;
  _Float16* qc  = ws + off; off += (size_t)NB * LQ * CDIM;
  _Float16* kc  = ws + off; off += (size_t)NB * LKV * CDIM;
  _Float16* vc  = ws + off; off += (size_t)NB * LKV * CDIM;
  _Float16* Wqh = ws + off; off += (size_t)CDIM * CDIM;
  _Float16* Wkh = ws + off; off += (size_t)CDIM * CDIM;
  _Float16* Wvh = ws + off; off += (size_t)CDIM * CDIM;
  _Float16* Qh  = ws + off; off += (size_t)NB * LQ * CDIM;
  _Float16* Kh  = ws + off; off += (size_t)NB * LKV * CDIM;
  _Float16* Vh  = ws + off; off += (size_t)NB * LKV * CDIM;

  cvt_w_kernel<<<(CDIM * CDIM + 255) / 256, 256, 0, stream>>>(Wq, Wk, Wv, Wqh, Wkh, Wvh);
  conv_bn_q_kernel<<<NB * LQ, CDIM, 0, stream>>>(hs, wdq, gq, beq, muq, vaq, qc);
  conv_bn_kv_kernel<<<NB * LKV, CDIM, 0, stream>>>(hs, wdk, gk, bek, muk, vak,
                                                   wdv, gv, bev, muv, vav, kc, vc);
  constexpr int MQ = NB * LQ, MKV = NB * LKV;
  proj_gemm_kernel<LQ><<<dim3((MQ + 127) / 128, CDIM / 64), 256, 0, stream>>>(qc, Wqh, bq, Qh);
  proj_gemm_kernel<LKV><<<dim3((MKV + 127) / 128, CDIM / 64), 256, 0, stream>>>(kc, Wkh, bk, Kh);
  proj_gemm_kernel<LKV><<<dim3((MKV + 127) / 128, CDIM / 64), 256, 0, stream>>>(vc, Wvh, bv, Vh);
  flash_attn_kernel<<<dim3(NB * NHEAD, QSPLIT), 256, 0, stream>>>(Qh, Kh, Vh, (float*)d_out);
}
